// GeometricTerm_82592221102153
// MI455X (gfx1250) — matrix operand order repaired and checked by reading, not compiled, not run on hardware
//
#include <hip/hip_runtime.h>
#include <stdint.h>

// ---------------- problem constants (from reference) ----------------
#define C_DIM   512                  // sinusoid embedding dim
#define NBOX    128                  // N rois
#define HDIM    64
#define WDIM    96
#define KTAP    49                   // 7x7
#define ODIM    256                  // output channel dim (GEMM K)
#define MROWS   (NBOX * KTAP)        // 6272  (GEMM M)
#define NCOLS   (HDIM * WDIM)        // 6144  (GEMM N)
#define BSTRIDE 264                  // padded LDS row stride (halves): 528B -> 4-bank rotation/row

typedef __attribute__((ext_vector_type(16))) _Float16 v16h;
typedef __attribute__((ext_vector_type(8)))  float    v8f;
typedef __attribute__((ext_vector_type(4)))  float    f32x4;   // 16-byte vector load

// sinusoid(z)[j], j in [0,512): j<256 -> sin(z*1000^{-(2j+1)/512}),
//                               j>=256 -> cos(z*1000^{-(2(j-256))/512})
__device__ __forceinline__ float sinusoid_val(float z, int j) {
  const float ln1000 = 6.907755278982137f;
  if (j < 256) {
    float e = (2.0f * (float)j + 1.0f) * (1.0f / 512.0f);
    return __sinf(z * __expf(-e * ln1000));
  } else {
    float e = (2.0f * (float)(j - 256)) * (1.0f / 512.0f);
    return __cosf(z * __expf(-e * ln1000));
  }
}

// ---------------- tiny prologue kernels (together ~20% of total flops) ----------------

__global__ void k_idx(const float* __restrict__ rois, float* __restrict__ out) {
  int n = threadIdx.x;
  if (n < NBOX) out[n] = rois[n * 5 + 0];
}

// eps_box[N,2048] fp32
__global__ void k_eps(const float* __restrict__ rois, float* __restrict__ eps) {
  int t = blockIdx.x * blockDim.x + threadIdx.x;     // 128*2048
  int n = t >> 11, d = t & 2047;
  int coord = d >> 9, j = d & 511;
  float z = rois[n * 5 + 1 + coord];
  eps[t] = sinusoid_val(z, j);
}

// v[N,512] = eps_box @ V_box^T
__global__ void k_v(const float* __restrict__ eps, const float* __restrict__ Vb,
                    float* __restrict__ v) {
  int t = blockIdx.x * blockDim.x + threadIdx.x;     // 128*512
  int n = t >> 9, c = t & 511;
  const float* er = eps + (size_t)n * 2048;
  const float* vr = Vb  + (size_t)c * 2048;
  float s = 0.f;
  #pragma unroll 8
  for (int d = 0; d < 2048; ++d) s = fmaf(er[d], vr[d], s);
  v[t] = s;
}

// A[M=6272, K=256] (f16) : A[(n*49+k)*256+o] = sum_c W_box[k,o,c]*v[n,c]
__global__ void k_box(const float* __restrict__ v, const float* __restrict__ Wb,
                      _Float16* __restrict__ A) {
  int t = blockIdx.x * blockDim.x + threadIdx.x;     // MROWS*256
  int row = t >> 8, o = t & 255;
  int n = row / KTAP, k = row % KTAP;
  const float* wr = Wb + ((size_t)k * 256 + o) * 512;
  const float* vr = v  + (size_t)n * 512;
  float s = 0.f;
  #pragma unroll 8
  for (int c = 0; c < 512; ++c) s = fmaf(wr[c], vr[c], s);
  A[t] = (_Float16)s;
}

// posTab rows 0..63 = sinusoid(h), rows 64..159 = sinusoid(w)
__global__ void k_pos(float* __restrict__ pos) {
  int t = blockIdx.x * blockDim.x + threadIdx.x;     // 160*512
  int r = t >> 9, j = t & 511;
  float z = (r < HDIM) ? (float)r : (float)(r - HDIM);
  pos[t] = sinusoid_val(z, j);
}

// B^T[Ncol=6144, K=256] (f16): Bm[hw*256+o] = sum_c W_im[o,c]*pos(h,w)[c]
__global__ void k_img(const float* __restrict__ pos, const float* __restrict__ Wim,
                      _Float16* __restrict__ Bm) {
  int t = blockIdx.x * blockDim.x + threadIdx.x;     // NCOLS*256
  int hw = t >> 8, o = t & 255;
  int h = hw / WDIM, w = hw % WDIM;
  const float* ph = pos + (size_t)h * 512;
  const float* pw = pos + (size_t)(HDIM + w) * 512;
  const float* wr = Wim + (size_t)o * 1024;
  float s = 0.f;
  #pragma unroll 8
  for (int c = 0; c < 512; ++c) s = fmaf(ph[c], wr[c], s);
  #pragma unroll 8
  for (int c = 0; c < 512; ++c) s = fmaf(pw[c], wr[512 + c], s);
  Bm[t] = (_Float16)s;
}

// ---------------- big GEMM: geo[6272,6144] = A(f16) x B(f16), f32 acc ----------------
// Block = 256 threads = 8 waves; block tile 128(M) x 64(N).
// B tile (64x256 f16) staged into LDS ONCE via async copy (ASYNCcnt path),
// padded rows (BSTRIDE=264 halves) for conflict-free ds_load_b128 fragment reads.
// Each wave: 16x64 output via 4 accumulators, A fragment reused 4x, K-loop 8x32.
__global__ void __launch_bounds__(256)
k_geo_wmma(const _Float16* __restrict__ A, const _Float16* __restrict__ Bm,
           float* __restrict__ D) {
  __shared__ _Float16 Bs[64 * BSTRIDE];              // 33,792 B

  const int lane = threadIdx.x & 31;
  const int wave = threadIdx.x >> 5;
  const int half = lane >> 4;          // 0 | 1
  const int l16  = lane & 15;
  const int nBase = blockIdx.x * 64;               // 96 blocks in x
  const int m0    = (blockIdx.y * 8 + wave) * 16;  // 49 blocks in y * 8 waves

  // ---- async-stage B tile: 64 rows x 256 halves = 2048 x 16B chunks ----
  {
    const unsigned ldsBase = (unsigned)(uintptr_t)(void*)&Bs[0]; // low 32 bits = LDS offset
    #pragma unroll
    for (int it = 0; it < 8; ++it) {
      const int i    = it * 256 + (int)threadIdx.x;  // chunk id
      const int row  = i >> 5;                       // 0..63
      const int chnk = i & 31;                       // 0..31 (16B each)
      const _Float16* g = Bm + (size_t)(nBase + row) * ODIM + chnk * 8;
      const unsigned  l = ldsBase + (unsigned)(row * BSTRIDE + chnk * 8) * 2u;
      asm volatile("global_load_async_to_lds_b128 %0, %1, off"
                   :: "v"(l), "v"(g) : "memory");
    }
    asm volatile("s_wait_asynccnt 0x0" ::: "memory");
  }
  __syncthreads();

  const _Float16* aRow = A + (size_t)(m0 + l16) * ODIM;
  __builtin_prefetch(aRow, 0, 1);                    // global_prefetch_b8

  v8f acc0 = {}, acc1 = {}, acc2 = {}, acc3 = {};
  const _Float16* bRow0 = &Bs[(  0 + l16) * BSTRIDE];
  const _Float16* bRow1 = &Bs[( 16 + l16) * BSTRIDE];
  const _Float16* bRow2 = &Bs[( 32 + l16) * BSTRIDE];
  const _Float16* bRow3 = &Bs[( 48 + l16) * BSTRIDE];

  #pragma unroll
  for (int k = 0; k < ODIM; k += 32) {
    union { v16h v; f32x4 q[2]; } af, b0, b1, b2, b3;
    const int ka = k + half * 16;      // A contiguous halves (matches B)
    af.q[0] = *(const f32x4*)(aRow + ka);
    af.q[1] = *(const f32x4*)(aRow + ka + 8);
    const int kb = k + half * 16;      // B contiguous halves (same k order as A)
    b0.q[0] = *(const f32x4*)(bRow0 + kb); b0.q[1] = *(const f32x4*)(bRow0 + kb + 8);
    b1.q[0] = *(const f32x4*)(bRow1 + kb); b1.q[1] = *(const f32x4*)(bRow1 + kb + 8);
    b2.q[0] = *(const f32x4*)(bRow2 + kb); b2.q[1] = *(const f32x4*)(bRow2 + kb + 8);
    b3.q[0] = *(const f32x4*)(bRow3 + kb); b3.q[1] = *(const f32x4*)(bRow3 + kb + 8);

    acc0 = __builtin_amdgcn_wmma_f32_16x16x32_f16(false, af.v, false, b0.v, (short)0, acc0, false, false);
    acc1 = __builtin_amdgcn_wmma_f32_16x16x32_f16(false, af.v, false, b1.v, (short)0, acc1, false, false);
    acc2 = __builtin_amdgcn_wmma_f32_16x16x32_f16(false, af.v, false, b2.v, (short)0, acc2, false, false);
    acc3 = __builtin_amdgcn_wmma_f32_16x16x32_f16(false, af.v, false, b3.v, (short)0, acc3, false, false);
  }

  // scatter D per the 16x16 f32 C/D layout: VGPR r, lane L -> row r+8*(L>=16), col L&15
  union { v8f v; float f[8]; } u;
  #pragma unroll
  for (int t = 0; t < 4; ++t) {
    u.v = (t == 0) ? acc0 : (t == 1) ? acc1 : (t == 2) ? acc2 : acc3;
    const int col = nBase + t * 16 + l16;
    #pragma unroll
    for (int r = 0; r < 8; ++r)
      D[(size_t)(m0 + r + 8 * half) * NCOLS + col] = u.f[r];
  }
}

// ---------------- launch ----------------
extern "C" void kernel_launch(void* const* d_in, const int* in_sizes, int n_in,
                              void* d_out, int out_size, void* d_ws, size_t ws_size,
                              hipStream_t stream) {
  // inputs: 0=input (unused by outputs), 1=rois, 2=V_box, 3=W_box, 4=W_im
  const float* rois = (const float*)d_in[1];
  const float* Vb   = (const float*)d_in[2];
  const float* Wb   = (const float*)d_in[3];
  const float* Wim  = (const float*)d_in[4];

  // workspace layout (bytes, 16B-aligned offsets), total ~7.6 MB
  char* ws = (char*)d_ws;
  float*    eps = (float*)(ws + 0);              // 128*2048*4 = 1,048,576
  float*    v   = (float*)(ws + 1048576);        // 128*512*4  =   262,144
  float*    pos = (float*)(ws + 1310720);        // 160*512*4  =   327,680
  _Float16* Ah  = (_Float16*)(ws + 1638400);     // 6272*256*2 = 3,211,264
  _Float16* Bh  = (_Float16*)(ws + 4849664);     // 6144*256*2 = 3,145,728

  float* idx_out = (float*)d_out;                // indices first, then geo
  float* geo     = (float*)d_out + NBOX;

  k_idx<<<1, 128, 0, stream>>>(rois, idx_out);
  k_eps<<<(NBOX * 2048) / 256, 256, 0, stream>>>(rois, eps);
  k_pos<<<(160 * 512) / 256, 256, 0, stream>>>(pos);
  k_v  <<<(NBOX * 512) / 256, 256, 0, stream>>>(eps, Vb, v);
  k_box<<<(MROWS * ODIM) / 256, 256, 0, stream>>>(v, Wb, Ah);
  k_img<<<(NCOLS * ODIM) / 256, 256, 0, stream>>>(pos, Wim, Bh);

  dim3 g(NCOLS / 64, MROWS / (16 * 8));          // 96 x 49 blocks
  k_geo_wmma<<<g, 256, 0, stream>>>(Ah, Bh, geo);
}
